// MultiRelationEncoder_20040317403569
// MI455X (gfx1250) — matrix operand order repaired and checked by reading, not compiled, not run on hardware
//
#include <hip/hip_runtime.h>
#include <hip/hip_bf16.h>
#include <math.h>

// Problem constants (from reference)
#define BDIM 16
#define NN   512
#define DD   256
#define HH   8
#define LL   4
#define DFFN 1024
#define KSEL 128
#define DHH  32
#define MROWS (BDIM*NN)   // 8192 rows in the flattened (B*N, D) activation

typedef __bf16 bf16;
typedef __attribute__((ext_vector_type(16))) __bf16 v16bf;
typedef __attribute__((ext_vector_type(8)))  __bf16 v8bf;
typedef __attribute__((ext_vector_type(8)))  float  v8f;
typedef __attribute__((ext_vector_type(4)))  unsigned int v4u;
typedef __attribute__((ext_vector_type(8)))  int v8i;
typedef __attribute__((ext_vector_type(4)))  int v4i;

__device__ __forceinline__ v16bf cat16(v8bf lo, v8bf hi) {
  return __builtin_shufflevector(lo, hi, 0,1,2,3,4,5,6,7,8,9,10,11,12,13,14,15);
}

__device__ __forceinline__ float gelu_tanh(float x) {
  float x3 = x * x * x;
  return 0.5f * x * (1.0f + tanhf(0.7978845608028654f * (x + 0.044715f * x3)));
}

// For LDS objects the low 32 bits of the flat address are the group-segment
// byte offset (ISA §10.2 aperture mapping). Taking the address here also makes
// the shared array escape, so async/TDM writes can't be folded away.
__device__ __forceinline__ unsigned lds_off(const void* p) {
  return (unsigned)(unsigned long long)p;
}

// ---------------------------------------------------------------------------
// Tensor Data Mover: 2D tile load Global -> LDS (CDNA5 TDM, ISA ch.8 D#).
// ldsPtr     : pointer into shared memory (destination of the tile)
// gaddr      : global address of tile start
// rowDwords  : tile/tensor dim0 in 4-byte units
// rows       : tile dim1 (rows)
// strideDw   : tensor dim0 stride in dwords
// tensorRows : tensor dim1 (for OOB checks)
// ---------------------------------------------------------------------------
__device__ __forceinline__ void tdm_load_2d(void* ldsPtr, const void* gaddr,
                                            unsigned rowDwords, unsigned rows,
                                            unsigned strideDw, unsigned tensorRows) {
  unsigned long long ga = (unsigned long long)gaddr;
  v4u g0;
  g0.x = 1u;                                         // count=1, user mode
  g0.y = lds_off(ldsPtr);                            // lds_addr (bytes)
  g0.z = (unsigned)(ga & 0xFFFFFFFFu);               // global_addr[31:0]
  g0.w = (unsigned)((ga >> 32) & 0x01FFFFFFu)        // global_addr[56:32]
       | (2u << 30);                                 // type=2 ("image")
  v8i g1;
  g1[0] = (int)(2u << 16);                           // wg_mask=0, data_size=2 (4B)
  g1[1] = (int)((rowDwords & 0xFFFFu) << 16);        // tensor_dim0[15:0]
  g1[2] = (int)((rowDwords >> 16) & 0xFFFFu)         // tensor_dim0[31:16]
        | (int)((tensorRows & 0xFFFFu) << 16);       // tensor_dim1[15:0]
  g1[3] = (int)((tensorRows >> 16) & 0xFFFFu)        // tensor_dim1[31:16]
        | (int)((rowDwords & 0xFFFFu) << 16);        // tile_dim0
  g1[4] = (int)(rows & 0xFFFFu);                     // tile_dim1 (tile_dim2=0)
  g1[5] = (int)strideDw;                             // tensor_dim0_stride[31:0]
  g1[6] = 0;                                         // stride hi / dim1_stride lo
  g1[7] = 0;                                         // dim1_stride hi
  v4i z4 = {0, 0, 0, 0};                             // groups 2/3 unused (2D)
  v8i z8 = {0, 0, 0, 0, 0, 0, 0, 0};
  // amdgpu-toolchain (clang-23) 6-arg form: (g0, g1, g2, g3, g4, cpol)
  __builtin_amdgcn_tensor_load_to_lds(g0, g1, z4, z4, z8, 0);
}

// ---------------------------------------------------------------------------
// One-time weight prep: f32 (R,C) row-major -> bf16 (C,R) row-major, per layer
// ---------------------------------------------------------------------------
__global__ void wprep_transpose_bf16(const float* __restrict__ W, bf16* __restrict__ Wt,
                                     int R, int C) {
  int l = blockIdx.y;
  size_t RC = (size_t)R * C;
  size_t t = (size_t)blockIdx.x * blockDim.x + threadIdx.x;
  if (t < RC) {
    int r = (int)(t / C), c = (int)(t % C);
    Wt[l*RC + (size_t)c*R + r] = (bf16)W[l*RC + t];
  }
}

// ---------------------------------------------------------------------------
// LayerNorm over D=256, one block per row, bf16 output (feeds WMMA A operand)
// ---------------------------------------------------------------------------
__global__ __launch_bounds__(DD) void layernorm_to_bf16(
    const float* __restrict__ x, const float* __restrict__ g,
    const float* __restrict__ b, bf16* __restrict__ out) {
  __shared__ float red[DD];
  const int row = blockIdx.x, t = threadIdx.x;
  float v = x[(size_t)row*DD + t];
  red[t] = v; __syncthreads();
  for (int off = 128; off > 0; off >>= 1) { if (t < off) red[t] += red[t+off]; __syncthreads(); }
  float mu = red[0] * (1.0f/DD); __syncthreads();
  float c = v - mu;
  red[t] = c*c; __syncthreads();
  for (int off = 128; off > 0; off >>= 1) { if (t < off) red[t] += red[t+off]; __syncthreads(); }
  float var = red[0] * (1.0f/DD);
  out[(size_t)row*DD + t] = (bf16)(c * rsqrtf(var + 1e-6f) * g[t] + b[t]);
}

// ---------------------------------------------------------------------------
// WMMA GEMM:  C[M,Ncols] = A[M,K] @ W[K,Ncols]. A-tile (16 x K bf16) staged to
// LDS by the Tensor Data Mover (one descriptor per block, issued by wave 0).
// MODE 0: store bf16 row-major            (Q, K projections)
// MODE 1: store bf16 transposed per head  (V -> Vt[B,H,DH,N])
// MODE 2: out_f32 = resid + acc           (O-proj, FFN2 residual adds)
// MODE 3: store bf16 gelu(acc)            (FFN1)
// ---------------------------------------------------------------------------
template<int K, int MODE>
__global__ __launch_bounds__(256) void gemm_wmma(
    const bf16* __restrict__ A, const bf16* __restrict__ Wt, int Ncols,
    const float* __restrict__ resid, float* __restrict__ out_f32,
    bf16* __restrict__ out_bf16) {
  __shared__ bf16 sA[16*K];
  const int tid  = threadIdx.x;
  const int row0 = blockIdx.x * 16;

  if (tid < 32) {   // wave 0 issues one TDM 2D tile load: 16 rows x K bf16
    tdm_load_2d(&sA[0], A + (size_t)row0*K, K/2, 16, K/2, MROWS);
  }
  __builtin_amdgcn_s_wait_tensorcnt(0);
  __syncthreads();

  const int wave = tid >> 5, lane = tid & 31;
  const int l15 = lane & 15, lhalf = lane >> 4;
  const int ncol = blockIdx.y*128 + wave*16;
  const bf16* abase = &sA[l15*K + lhalf*16];
  const bf16* bbase = Wt + (size_t)(ncol + l15)*K + lhalf*16;

  v8f acc = {};
  #pragma unroll
  for (int k0 = 0; k0 < K; k0 += 32) {
    __builtin_prefetch((const void*)(bbase + k0 + 64), 0, 1);   // global_prefetch_b8
    v8bf alo = *(const v8bf*)(abase + k0);
    v8bf ahi = *(const v8bf*)(abase + k0 + 8);
    v8bf blo = *(const v8bf*)(bbase + k0);
    v8bf bhi = *(const v8bf*)(bbase + k0 + 8);
    acc = __builtin_amdgcn_wmma_f32_16x16x32_bf16(false, cat16(alo, ahi),
                                                  false, cat16(blo, bhi),
                                                  (short)0, acc, false, false);
  }

  const int col = ncol + l15;
  const int rbase = row0 + lhalf*8;
  #pragma unroll
  for (int r = 0; r < 8; ++r) {
    const int row = rbase + r;
    float v = acc[r];
    if (MODE == 0) {
      out_bf16[(size_t)row*Ncols + col] = (bf16)v;
    } else if (MODE == 1) {
      int b = row >> 9, node = row & 511;
      int h = col >> 5, dh = col & 31;
      out_bf16[(((size_t)(b*HH + h))*DHH + dh)*NN + node] = (bf16)v;
    } else if (MODE == 2) {
      size_t o = (size_t)row*Ncols + col;
      out_f32[o] = resid[o] + v;
    } else {
      out_bf16[(size_t)row*Ncols + col] = (bf16)gelu_tanh(v);
    }
  }
}

// ---------------------------------------------------------------------------
// Attention logits: per (b,h), logits = (Q K^T)/sqrt(DH) - dist, stored bf16.
// Q-tile (16x32 bf16, 1KB) staged via async global->LDS (ASYNCcnt path).
// ---------------------------------------------------------------------------
__global__ __launch_bounds__(256) void attn_logits_wmma(
    const bf16* __restrict__ Q, const bf16* __restrict__ Kmat,
    const float* __restrict__ dist, bf16* __restrict__ logits) {
  __shared__ bf16 sQ[16*DHH];
  const int tid = threadIdx.x;
  const int bh = blockIdx.z, b = bh >> 3, h = bh & 7;
  const int q0 = blockIdx.x * 16;
  if (tid < 64) {
    int rr = tid >> 2, cc = (tid & 3) * 8;
    // real LDS pointer -> escapes sQ; low 32 bits = group-segment offset
    unsigned ldsOff = lds_off(&sQ[rr*DHH + cc]);
    unsigned long long ga =
        (unsigned long long)(size_t)&Q[((size_t)(b*NN + q0 + rr))*DD + h*DHH + cc];
    asm volatile("global_load_async_to_lds_b128 %0, %1, off"
                 :: "v"(ldsOff), "v"(ga) : "memory");
  }
  asm volatile("s_wait_asynccnt 0x0" ::: "memory");
  __syncthreads();

  const int wave = tid >> 5, lane = tid & 31;
  const int l15 = lane & 15, lhalf = lane >> 4;
  const int kt = (blockIdx.y*8 + wave)*16;

  const bf16* ap = &sQ[l15*DHH + lhalf*16];
  v8bf alo = *(const v8bf*)ap;
  v8bf ahi = *(const v8bf*)(ap + 8);
  const bf16* bp = &Kmat[((size_t)(b*NN + kt + l15))*DD + h*DHH + lhalf*16];
  v8bf blo = *(const v8bf*)bp;
  v8bf bhi = *(const v8bf*)(bp + 8);

  v8f acc = {};
  acc = __builtin_amdgcn_wmma_f32_16x16x32_bf16(false, cat16(alo, ahi),
                                                false, cat16(blo, bhi),
                                                (short)0, acc, false, false);
  const float inv_sqrt = 0.17677669529663687f;  // 1/sqrt(32)
  const int kcol = kt + l15;
  #pragma unroll
  for (int r = 0; r < 8; ++r) {
    int qr = q0 + lhalf*8 + r;
    float v = acc[r]*inv_sqrt - dist[((size_t)b*NN + qr)*NN + kcol];
    logits[(((size_t)bh)*NN + qr)*NN + kcol] = (bf16)v;
  }
}

// In-place row softmax on bf16 logits -> bf16 scores (mask is all-true).
__global__ __launch_bounds__(256) void softmax_rows(bf16* __restrict__ sc) {
  __shared__ float red[256];
  const size_t row = blockIdx.x;
  bf16* p = sc + row*NN;
  const int t = threadIdx.x;
  float v0 = (float)p[t], v1 = (float)p[t+256];
  red[t] = fmaxf(v0, v1); __syncthreads();
  for (int off = 128; off > 0; off >>= 1) { if (t < off) red[t] = fmaxf(red[t], red[t+off]); __syncthreads(); }
  float m = red[0]; __syncthreads();
  float e0 = __expf(v0 - m), e1 = __expf(v1 - m);
  red[t] = e0 + e1; __syncthreads();
  for (int off = 128; off > 0; off >>= 1) { if (t < off) red[t] += red[t+off]; __syncthreads(); }
  float inv = 1.0f / red[0];
  p[t] = (bf16)(e0*inv); p[t+256] = (bf16)(e1*inv);
}

// attn = scores @ V, per (b,h); output written into AO[B*N, D] bf16 (head-merged).
__global__ __launch_bounds__(256) void attn_pv_wmma(
    const bf16* __restrict__ sc, const bf16* __restrict__ Vt, bf16* __restrict__ AO) {
  const int tid = threadIdx.x;
  const int wave = tid >> 5, lane = tid & 31;
  const int l15 = lane & 15, lhalf = lane >> 4;
  const int bh = blockIdx.y, b = bh >> 3, h = bh & 7;
  const int q0 = (blockIdx.x*4 + (wave >> 1))*16;
  const int dt = (wave & 1)*16;
  const bf16* ab = sc + ((size_t)bh*NN + q0 + l15)*NN + lhalf*16;
  const bf16* bb = Vt + ((size_t)(b*HH + h)*DHH + dt + l15)*NN + lhalf*16;
  v8f acc = {};
  #pragma unroll
  for (int k0 = 0; k0 < NN; k0 += 32) {
    v8bf alo = *(const v8bf*)(ab + k0);
    v8bf ahi = *(const v8bf*)(ab + k0 + 8);
    v8bf blo = *(const v8bf*)(bb + k0);
    v8bf bhi = *(const v8bf*)(bb + k0 + 8);
    acc = __builtin_amdgcn_wmma_f32_16x16x32_bf16(false, cat16(alo, ahi),
                                                  false, cat16(blo, bhi),
                                                  (short)0, acc, false, false);
  }
  #pragma unroll
  for (int r = 0; r < 8; ++r) {
    int qr = q0 + lhalf*8 + r;
    AO[((size_t)(b*NN + qr))*DD + h*DHH + dt + l15] = (bf16)acc[r];
  }
}

// s[b,n] = (1/H) * sum_{h,q} scores[b,h,q,n]  (coalesced across n)
__global__ __launch_bounds__(NN) void col_sums_kernel(
    const bf16* __restrict__ scores, float* __restrict__ s) {
  const int b = blockIdx.x, t = threadIdx.x;
  const bf16* p = scores + (size_t)b*HH*NN*NN;
  float acc = 0.0f;
  for (int r = 0; r < HH*NN; ++r) acc += (float)p[(size_t)r*NN + t];
  s[b*NN + t] = acc * (1.0f/HH);
}

// Sequential attention-biased farthest point sampling: one block per batch.
__global__ __launch_bounds__(NN) void afps_kernel(
    const float* __restrict__ dist, const float* __restrict__ s, int* __restrict__ idx) {
  __shared__ float red[NN];
  __shared__ int   redi[NN];
  const int b = blockIdx.x, t = threadIdx.x;
  const float* db = dist + (size_t)b*NN*NN;
  const float* rowp = db + (size_t)t*NN;

  float m = -1e30f;
  for (int j = 0; j < NN; ++j) m = fmaxf(m, rowp[j]);
  red[t] = m; __syncthreads();
  for (int off = 256; off > 0; off >>= 1) { if (t < off) red[t] = fmaxf(red[t], red[t+off]); __syncthreads(); }
  float maxd = red[0]; __syncthreads();

  float sv = s[b*NN + t];
  red[t] = sv; __syncthreads();
  for (int off = 256; off > 0; off >>= 1) { if (t < off) red[t] = fmaxf(red[t], red[t+off]); __syncthreads(); }
  float maxs = red[0]; __syncthreads();

  const float invd = 1.0f / maxd;
  const float sterm = sv * 0.1f / maxs;
  float mind = rowp[0]*invd + sterm;   // d[:,0]
  bool sel = (t == 0);
  if (t == 0) idx[b*KSEL] = 0;

  for (int it = 1; it < KSEL; ++it) {
    red[t]  = sel ? -1e38f : mind;
    redi[t] = t;
    __syncthreads();
    for (int off = 256; off > 0; off >>= 1) {
      if (t < off) {
        float a0 = red[t], a1 = red[t+off];
        int   i0 = redi[t], i1 = redi[t+off];
        if (a1 > a0 || (a1 == a0 && i1 < i0)) { red[t] = a1; redi[t] = i1; }
      }
      __syncthreads();
    }
    int nw = redi[0]; __syncthreads();
    if (t == nw) sel = true;
    if (t == 0) idx[b*KSEL + it] = nw;
    mind = fminf(mind, rowp[nw]*invd + sterm);   // d[:, new]
  }
}

// pooled = mean over selected rows of x; out = LN(pooled) with gamma/beta
__global__ __launch_bounds__(DD) void pool_ln_kernel(
    const float* __restrict__ x, const int* __restrict__ idx,
    const float* __restrict__ g, const float* __restrict__ bta, float* __restrict__ out) {
  __shared__ float red[DD];
  const int b = blockIdx.x, t = threadIdx.x;
  float acc = 0.0f;
  for (int i = 0; i < KSEL; ++i)
    acc += x[((size_t)b*NN + idx[b*KSEL + i])*DD + t];
  float p = acc * (1.0f/KSEL);
  red[t] = p; __syncthreads();
  for (int off = 128; off > 0; off >>= 1) { if (t < off) red[t] += red[t+off]; __syncthreads(); }
  float mu = red[0] * (1.0f/DD); __syncthreads();
  float c = p - mu;
  red[t] = c*c; __syncthreads();
  for (int off = 128; off > 0; off >>= 1) { if (t < off) red[t] += red[t+off]; __syncthreads(); }
  float var = red[0] * (1.0f/DD);
  out[b*DD + t] = c * rsqrtf(var + 1e-6f) * g[t] + bta[t];
}

// ---------------------------------------------------------------------------
extern "C" void kernel_launch(void* const* d_in, const int* in_sizes, int n_in,
                              void* d_out, int out_size, void* d_ws, size_t ws_size,
                              hipStream_t stream) {
  (void)in_sizes; (void)n_in; (void)out_size; (void)ws_size;
  const float* x    = (const float*)d_in[0];
  const float* dist = (const float*)d_in[1];
  // d_in[2] = mask (all true) — folded out
  const float* Wq   = (const float*)d_in[3];
  const float* Wk   = (const float*)d_in[4];
  const float* Wv   = (const float*)d_in[5];
  const float* Wo   = (const float*)d_in[6];
  const float* W1   = (const float*)d_in[7];
  const float* W2   = (const float*)d_in[8];
  const float* ln1g = (const float*)d_in[9];
  const float* ln1b = (const float*)d_in[10];
  const float* ln2g = (const float*)d_in[11];
  const float* ln2b = (const float*)d_in[12];
  const float* gamma= (const float*)d_in[13];
  const float* beta = (const float*)d_in[14];
  float* out = (float*)d_out;

  char* p = (char*)d_ws;
  auto take = [&p](size_t bytes) -> char* {
    char* r = p; p += (bytes + 255) & ~(size_t)255; return r;
  };
  bf16* WqT  = (bf16*)take((size_t)LL*DD*DD*sizeof(bf16));
  bf16* WkT  = (bf16*)take((size_t)LL*DD*DD*sizeof(bf16));
  bf16* WvT  = (bf16*)take((size_t)LL*DD*DD*sizeof(bf16));
  bf16* WoT  = (bf16*)take((size_t)LL*DD*DD*sizeof(bf16));
  bf16* W1T  = (bf16*)take((size_t)LL*DD*DFFN*sizeof(bf16));
  bf16* W2T  = (bf16*)take((size_t)LL*DD*DFFN*sizeof(bf16));
  bf16* hb   = (bf16*)take((size_t)MROWS*DD*sizeof(bf16));
  bf16* Qb   = (bf16*)take((size_t)MROWS*DD*sizeof(bf16));
  bf16* Kb   = (bf16*)take((size_t)MROWS*DD*sizeof(bf16));
  bf16* Vtb  = (bf16*)take((size_t)BDIM*HH*DHH*NN*sizeof(bf16));
  bf16* AOb  = (bf16*)take((size_t)MROWS*DD*sizeof(bf16));
  bf16* Gb   = (bf16*)take((size_t)MROWS*DFFN*sizeof(bf16));
  bf16* scores = (bf16*)take((size_t)BDIM*HH*NN*NN*sizeof(bf16));
  float* xw  = (float*)take((size_t)MROWS*DD*sizeof(float));
  float* sbuf= (float*)take((size_t)BDIM*NN*sizeof(float));
  int*  idxb = (int*) take((size_t)BDIM*KSEL*sizeof(int));

  // residual stream working copy (never mutate inputs)
  (void)hipMemcpyAsync(xw, x, (size_t)MROWS*DD*sizeof(float),
                       hipMemcpyDeviceToDevice, stream);

  // one-time weight transpose + f32->bf16
  wprep_transpose_bf16<<<dim3(DD*DD/256,  LL), 256, 0, stream>>>(Wq, WqT, DD, DD);
  wprep_transpose_bf16<<<dim3(DD*DD/256,  LL), 256, 0, stream>>>(Wk, WkT, DD, DD);
  wprep_transpose_bf16<<<dim3(DD*DD/256,  LL), 256, 0, stream>>>(Wv, WvT, DD, DD);
  wprep_transpose_bf16<<<dim3(DD*DD/256,  LL), 256, 0, stream>>>(Wo, WoT, DD, DD);
  wprep_transpose_bf16<<<dim3(DD*DFFN/256,LL), 256, 0, stream>>>(W1, W1T, DD,   DFFN);
  wprep_transpose_bf16<<<dim3(DD*DFFN/256,LL), 256, 0, stream>>>(W2, W2T, DFFN, DD);

  for (int i = 0; i < LL; ++i) {
    layernorm_to_bf16<<<MROWS, DD, 0, stream>>>(xw, ln1g + i*DD, ln1b + i*DD, hb);
    gemm_wmma<DD,0><<<dim3(MROWS/16, DD/128), 256, 0, stream>>>(
        hb, WqT + (size_t)i*DD*DD, DD, nullptr, nullptr, Qb);
    gemm_wmma<DD,0><<<dim3(MROWS/16, DD/128), 256, 0, stream>>>(
        hb, WkT + (size_t)i*DD*DD, DD, nullptr, nullptr, Kb);
    gemm_wmma<DD,1><<<dim3(MROWS/16, DD/128), 256, 0, stream>>>(
        hb, WvT + (size_t)i*DD*DD, DD, nullptr, nullptr, Vtb);

    attn_logits_wmma<<<dim3(NN/16, NN/128, BDIM*HH), 256, 0, stream>>>(Qb, Kb, dist, scores);
    softmax_rows<<<BDIM*HH*NN, 256, 0, stream>>>(scores);
    attn_pv_wmma<<<dim3(NN/64, BDIM*HH), 256, 0, stream>>>(scores, Vtb, AOb);

    gemm_wmma<DD,2><<<dim3(MROWS/16, DD/128), 256, 0, stream>>>(
        AOb, WoT + (size_t)i*DD*DD, DD, xw, xw, nullptr);

    layernorm_to_bf16<<<MROWS, DD, 0, stream>>>(xw, ln2g + i*DD, ln2b + i*DD, hb);
    gemm_wmma<DD,3><<<dim3(MROWS/16, DFFN/128), 256, 0, stream>>>(
        hb, W1T + (size_t)i*DD*DFFN, DFFN, nullptr, nullptr, Gb);
    gemm_wmma<DFFN,2><<<dim3(MROWS/16, DD/128), 256, 0, stream>>>(
        Gb, W2T + (size_t)i*DD*DFFN, DD, xw, xw, nullptr);
  }

  col_sums_kernel<<<BDIM, NN, 0, stream>>>(scores, sbuf);
  afps_kernel<<<BDIM, NN, 0, stream>>>(dist, sbuf, idxb);
  pool_ln_kernel<<<BDIM, DD, 0, stream>>>(xw, idxb, gamma, beta, out);
}
